// CrossVoxAttentionL2_58016418235045
// MI455X (gfx1250) — matrix operand order repaired and checked by reading, not compiled, not run on hardware
//
#include <hip/hip_runtime.h>

#define Bb   4
#define Vv   64
#define Ee   64
#define Hh   8
#define DSCc 256
#define SDd  1024
#define NPp  4096
#define LN_EPS 1e-5f
#define SCALEf 0.125f   // 1/sqrt(64)

typedef __attribute__((ext_vector_type(16))) _Float16 v16h;
typedef __attribute__((ext_vector_type(8)))  float    v8f;
typedef __attribute__((ext_vector_type(4)))  float    f4;

union FragH { v16h v; _Float16 h[16]; f4 q[2]; unsigned int u[8]; };
union AccF  { v8f  v; float f[8];    f4 q[2]; };
union Pack8 { f4   q; _Float16 h[8]; unsigned int u[4]; };

__device__ __forceinline__ int lane_id() { return (int)(threadIdx.x & 31u); }

// A fragment (M=rows): lane holds row rbase + lane%16.
// halves i = K = kbase + 16*(lane/16) + i
__device__ __forceinline__ v16h frag_a_h(const _Float16* base, int ld, int rbase, int kbase) {
  int l = lane_id(); int g = l >> 4;
  const _Float16* p = base + (size_t)(rbase + (l & 15)) * ld + kbase + 16 * g;
  FragH f;
  f.q[0] = *(const f4*)(p);
  f.q[1] = *(const f4*)(p + 8);
  return f.v;
}

// B fragment (N=cols), read from row-major [col][K] buffer (i.e. B^T):
// lane holds col cbase + lane%16; halves i = K = kbase + 16*(lane/16) + i
__device__ __forceinline__ v16h frag_b_h(const _Float16* base, int ld, int cbase, int kbase) {
  int l = lane_id(); int g = l >> 4;
  const _Float16* p = base + (size_t)(cbase + (l & 15)) * ld + kbase + 16 * g;
  FragH f;
  f.q[0] = *(const f4*)(p);
  f.q[1] = *(const f4*)(p + 8);
  return f.v;
}

// B fragment converting from fp32 source on the fly
__device__ __forceinline__ v16h frag_b_f32(const float* base, int ld, int cbase, int kbase) {
  int l = lane_id(); int g = l >> 4;
  const float* p = base + (size_t)(cbase + (l & 15)) * ld + kbase + 16 * g;
  FragH f;
#pragma unroll
  for (int i = 0; i < 16; ++i) f.h[i] = (_Float16)p[i];
  return f.v;
}

__device__ __forceinline__ v8f wmma_f16(v16h a, v16h b, v8f c) {
  return __builtin_amdgcn_wmma_f32_16x16x32_f16(false, a, false, b, (short)0, c, false, false);
}

__device__ __forceinline__ unsigned int pack2h(float a, float b) {
  union { unsigned int u; _Float16 h[2]; } t;
  t.h[0] = (_Float16)a; t.h[1] = (_Float16)b;
  return t.u;
}

// ---- CDNA5 async global->LDS (ASYNCcnt) --------------------------------
// LDS aperture keeps the byte offset in addr[31:0] (ISA 10.2), so truncating
// the flat shared address yields the LDS address operand.
__device__ __forceinline__ unsigned int lds_off(const void* p) {
  return (unsigned int)(unsigned long long)p;
}
__device__ __forceinline__ void async_copy_b128(unsigned int dst_lds, const void* src) {
  unsigned long long ga = (unsigned long long)src;
  asm volatile("global_load_async_to_lds_b128 %0, %1, off"
               :: "v"(dst_lds), "v"(ga) : "memory");
}
__device__ __forceinline__ void wait_async0() {
  asm volatile("s_wait_asynccnt 0x0" ::: "memory");
}

// ---------------------------------------------------------------------------
// 1) weight transpose + f32->f16: src [R][Cn] -> dst [Cn][R]
__global__ void k_transpose(const float* __restrict__ src, _Float16* __restrict__ dst,
                            int R, int Cn) {
  int i = blockIdx.x * blockDim.x + threadIdx.x;
  if (i >= R * Cn) return;
  int r = i / Cn, c = i - r * Cn;
  dst[c * R + r] = (_Float16)src[i];
}

// ---------------------------------------------------------------------------
// 2) patch embedding (64x64 mini-GEMM per patch) + LayerNorm + pos_emb -> f16
__global__ __launch_bounds__(64) void k_patch_embed(
    const float* __restrict__ vox, const float* __restrict__ cw,
    const float* __restrict__ cb,  const float* __restrict__ lng,
    const float* __restrict__ lnb, const float* __restrict__ pose,
    _Float16* __restrict__ vemb)
{
  __shared__ float patch[64];
  __shared__ float s1[64];
  __shared__ float s2[64];
  int p = blockIdx.x;                 // 0..16383  (b*4096 + pl)
  int b = p >> 12, pl = p & 4095;
  int pz = pl >> 8, py = (pl >> 4) & 15, px = pl & 15;
  int t = threadIdx.x;                // t = output channel E AND voxel idx
  int dz = t >> 4, dy = (t >> 2) & 3, dx = t & 3;
  patch[t] = vox[(((size_t)b * 64 + pz * 4 + dz) * 64 + (py * 4 + dy)) * 64 + (px * 4 + dx)];
  __syncthreads();
  float acc = cb[t];
#pragma unroll 8
  for (int i = 0; i < 64; ++i) acc += patch[i] * cw[t * 64 + i];
  s1[t] = acc; s2[t] = acc * acc;
  __syncthreads();
  for (int off = 32; off > 0; off >>= 1) {
    if (t < off) { s1[t] += s1[t + off]; s2[t] += s2[t + off]; }
    __syncthreads();
  }
  float mu  = s1[0] * (1.f / 64.f);
  float var = s2[0] * (1.f / 64.f) - mu * mu;
  float x = (acc - mu) * rsqrtf(var + LN_EPS) * lng[t] + lnb[t] + pose[pl * 64 + t];
  vemb[(size_t)p * 64 + t] = (_Float16)x;
}

// ---------------------------------------------------------------------------
// 3) des_emb[row][e] = descriptor[row][:] @ wd + bd   (row = b*1024+s, R=256)
__global__ __launch_bounds__(128) void k_desemb(
    const float* __restrict__ desc, const _Float16* __restrict__ wdT,
    const float* __restrict__ bd, _Float16* __restrict__ des)
{
  int wave = blockIdx.x * 4 + (threadIdx.x >> 5);   // 256 waves
  int rowbase = wave * 16;
  int l = lane_id(); int g = l >> 4; int nl = l & 15;
  v16h bf[8];
#pragma unroll
  for (int rb = 0; rb < 8; ++rb) bf[rb] = frag_b_f32(desc, DSCc, rowbase, rb * 32);
  _Float16* drow = des + (size_t)(rowbase + nl) * Ee + 8 * g;
#pragma unroll
  for (int t = 0; t < 4; ++t) {
    AccF c;
#pragma unroll
    for (int r = 0; r < 8; ++r) c.f[r] = 0.f;
#pragma unroll
    for (int rb = 0; rb < 8; ++rb)
      c.v = wmma_f16(frag_a_h(wdT, DSCc, t * 16, rb * 32), bf[rb], c.v);
    const float* bp = bd + t * 16 + 8 * g;
    Pack8 op;
#pragma unroll
    for (int r = 0; r < 8; ++r) op.h[r] = (_Float16)(c.f[r] + bp[r]);
    *(f4*)(drow + t * 16) = op.q;
  }
}

// ---------------------------------------------------------------------------
// 4) Q projection (+|q|^2): A = wqT (M=out), B = vox_emb (N=queries)
__global__ __launch_bounds__(128) void k_qproj(
    const _Float16* __restrict__ vemb, const _Float16* __restrict__ wqT,
    const float* __restrict__ bq, _Float16* __restrict__ Qh, float* __restrict__ q2)
{
  int wave = blockIdx.x * 4 + (threadIdx.x >> 5);   // 8192 waves
  int bh = wave >> 8; int qt = wave & 255;
  int b = bh >> 3, h = bh & 7;
  int qbase = qt * 16;
  int l = lane_id(); int g = l >> 4; int nl = l & 15;
  const _Float16* vb = vemb + (size_t)b * NPp * Ee;
  v16h b0 = frag_b_h(vb, Ee, qbase, 0);
  v16h b1 = frag_b_h(vb, Ee, qbase, 32);
  float q2part = 0.f;
  _Float16* qrow = Qh + ((size_t)bh * NPp + qbase + nl) * Ee + 8 * g;
#pragma unroll
  for (int t = 0; t < 4; ++t) {
    AccF c;
#pragma unroll
    for (int r = 0; r < 8; ++r) c.f[r] = 0.f;
    c.v = wmma_f16(frag_a_h(wqT, Ee, h * 64 + t * 16, 0),  b0, c.v);
    c.v = wmma_f16(frag_a_h(wqT, Ee, h * 64 + t * 16, 32), b1, c.v);
    const float* bp = bq + h * 64 + t * 16 + 8 * g;
    Pack8 op;
#pragma unroll
    for (int r = 0; r < 8; ++r) {
      float v = c.f[r] + bp[r];
      q2part += v * v;
      op.h[r] = (_Float16)v;
    }
    *(f4*)(qrow + t * 16) = op.q;
  }
  float q2f = q2part + __shfl_xor(q2part, 16, 32);
  if (l < 16) q2[(size_t)bh * NPp + qbase + nl] = q2f;
}

// ---------------------------------------------------------------------------
// 5) K/V projection from des_emb. K -> [bh][key][e] (+k2), V -> VT [bh][e][key]
__global__ __launch_bounds__(128) void k_kvproj(
    const _Float16* __restrict__ des, const _Float16* __restrict__ wkT,
    const float* __restrict__ bk, const _Float16* __restrict__ wvT,
    const float* __restrict__ bv, _Float16* __restrict__ Kh,
    float* __restrict__ k2, _Float16* __restrict__ VTh)
{
  int wave = blockIdx.x * 4 + (threadIdx.x >> 5);   // 2048 waves
  int bh = wave >> 6; int kt = wave & 63;
  int b = bh >> 3, h = bh & 7;
  int kbase = kt * 16;
  int l = lane_id(); int g = l >> 4; int nl = l & 15;
  const _Float16* db = des + (size_t)b * SDd * Ee;

  v16h b0 = frag_b_h(db, Ee, kbase, 0);
  v16h b1 = frag_b_h(db, Ee, kbase, 32);
  float k2part = 0.f;
  _Float16* krow = Kh + ((size_t)bh * SDd + kbase + nl) * Ee + 8 * g;
#pragma unroll
  for (int t = 0; t < 4; ++t) {
    AccF c;
#pragma unroll
    for (int r = 0; r < 8; ++r) c.f[r] = 0.f;
    c.v = wmma_f16(frag_a_h(wkT, Ee, h * 64 + t * 16, 0),  b0, c.v);
    c.v = wmma_f16(frag_a_h(wkT, Ee, h * 64 + t * 16, 32), b1, c.v);
    const float* bp = bk + h * 64 + t * 16 + 8 * g;
    Pack8 op;
#pragma unroll
    for (int r = 0; r < 8; ++r) {
      float v = c.f[r] + bp[r];
      k2part += v * v;
      op.h[r] = (_Float16)v;
    }
    *(f4*)(krow + t * 16) = op.q;
  }
  float k2f = k2part + __shfl_xor(k2part, 16, 32);
  if (l < 16) k2[(size_t)bh * SDd + kbase + nl] = k2f;

  v16h a0 = frag_a_h(db, Ee, kbase, 0);
  v16h a1 = frag_a_h(db, Ee, kbase, 32);
#pragma unroll
  for (int t = 0; t < 4; ++t) {
    AccF c;
#pragma unroll
    for (int r = 0; r < 8; ++r) c.f[r] = 0.f;
    c.v = wmma_f16(a0, frag_b_h(wvT, Ee, h * 64 + t * 16, 0),  c.v);
    c.v = wmma_f16(a1, frag_b_h(wvT, Ee, h * 64 + t * 16, 32), c.v);
    float bvv = bv[h * 64 + t * 16 + nl];
    Pack8 op;
#pragma unroll
    for (int r = 0; r < 8; ++r) op.h[r] = (_Float16)(c.f[r] + bvv);
    *(f4*)(VTh + ((size_t)bh * Ee + t * 16 + nl) * SDd + kbase + 8 * g) = op.q;
  }
}

// ---------------------------------------------------------------------------
// 6) attention with async double-buffered K/V staging in LDS.
//    Block = 4 waves sharing (b,h); each wave owns one 16-query tile.
//    Per 32-key block: K tile (32x64 f16, 4KB), V^T slice (64x32 f16, 4KB)
//    and k2 (32 f32) staged via global_load_async_to_lds_b128 (ASYNCcnt),
//    prefetching block n+1 while WMMAs consume block n from LDS.
__global__ __launch_bounds__(128) void k_attn(
    const _Float16* __restrict__ Qh, const float* __restrict__ q2,
    const _Float16* __restrict__ Kh, const float* __restrict__ k2,
    const _Float16* __restrict__ VTh, const float* __restrict__ alpha_p,
    _Float16* __restrict__ attn)
{
  __shared__ _Float16 Kst[2][32 * 64];   // [buf][key][e]
  __shared__ _Float16 Vst[2][64 * 32];   // [buf][e][key]
  __shared__ float    k2st[2][32];

  int bh = blockIdx.x >> 6;             // 32 (b,h) pairs
  int qg = blockIdx.x & 63;             // 64 query-groups of 64
  int wid = threadIdx.x >> 5;
  int tid = threadIdx.x;
  int b = bh >> 3, h = bh & 7;
  int qbase = qg * 64 + wid * 16;
  int l = lane_id(); int g = l >> 4; int nl = l & 15;

  const _Float16* Qb = Qh  + (size_t)bh * NPp * Ee;
  const _Float16* Kb = Kh  + (size_t)bh * SDd * Ee;
  const _Float16* Vb = VTh + (size_t)bh * Ee * SDd;
  const float*   k2b = k2  + (size_t)bh * SDd;
  float ascale = alpha_p[0] * SCALEf;

  v16h qb0 = frag_b_h(Qb, Ee, qbase, 0);
  v16h qb1 = frag_b_h(Qb, Ee, qbase, 32);
  float q2v = q2[(size_t)bh * NPp + qbase + nl];

  AccF acc[4];
#pragma unroll
  for (int t = 0; t < 4; ++t)
#pragma unroll
    for (int r = 0; r < 8; ++r) acc[t].f[r] = 0.f;
  float lpart = 0.f;

  // cooperative async stage of one 32-key block into buffer `buf`
  auto stage = [&](int buf, int kb) {
    // K tile: 4KB contiguous
    {
      const char* src = (const char*)(Kb + (size_t)kb * Ee);
      unsigned int dst = lds_off(&Kst[buf][0]);
      async_copy_b128(dst + tid * 16, src + tid * 16);
      async_copy_b128(dst + 2048 + tid * 16, src + 2048 + tid * 16);
    }
    // V^T slice: 64 rows x 64B (row stride SDd*2 bytes in global)
    {
      unsigned int dst = lds_off(&Vst[buf][0]);
      int c0 = tid, c1 = tid + 128;
      int r0 = c0 >> 2, o0 = (c0 & 3) * 16;
      int r1 = c1 >> 2, o1 = (c1 & 3) * 16;
      async_copy_b128(dst + r0 * 64 + o0, (const char*)(Vb + (size_t)r0 * SDd + kb) + o0);
      async_copy_b128(dst + r1 * 64 + o1, (const char*)(Vb + (size_t)r1 * SDd + kb) + o1);
    }
    // k2: 128B
    if (tid < 8)
      async_copy_b128(lds_off(&k2st[buf][0]) + tid * 16, (const char*)(k2b + kb) + tid * 16);
  };

  stage(0, 0);
  int cur = 0;
  for (int kb = 0; kb < SDd; kb += 32) {
    wait_async0();
    __syncthreads();
    if (kb + 32 < SDd) stage(cur ^ 1, kb + 32);

    const _Float16* Kl = &Kst[cur][0];
    const _Float16* Vl = &Vst[cur][0];
    AccF s1, s2;
#pragma unroll
    for (int r = 0; r < 8; ++r) { s1.f[r] = 0.f; s2.f[r] = 0.f; }
    // S^T tiles: M=key, N=query ; accumulate over E=64 in two K=32 steps
    s1.v = wmma_f16(frag_a_h(Kl, Ee, 0,  0),  qb0, s1.v);
    s1.v = wmma_f16(frag_a_h(Kl, Ee, 0,  32), qb1, s1.v);
    s2.v = wmma_f16(frag_a_h(Kl, Ee, 16, 0),  qb0, s2.v);
    s2.v = wmma_f16(frag_a_h(Kl, Ee, 16, 32), qb1, s2.v);

    const float* kp1 = &k2st[cur][8 * g];
    const float* kp2 = &k2st[cur][16 + 8 * g];
#pragma unroll
    for (int r = 0; r < 8; ++r) {
      float d1 = q2v + kp1[r] - 2.f * s1.f[r];   // |q-k|^2
      float d2 = q2v + kp2[r] - 2.f * s2.f[r];
      float p1 = __expf(ascale * __expf(-d1));   // softmax arg <= alpha/8
      float p2 = __expf(ascale * __expf(-d2));
      lpart += p1 + p2;
      s1.f[r] = p1; s2.f[r] = p2;
    }
    // build f16 B-fragment of P^T (32k x 16q) from the two C-tiles
    unsigned int d1p[4], d2p[4];
#pragma unroll
    for (int j = 0; j < 4; ++j) {
      d1p[j] = pack2h(s1.f[2 * j], s1.f[2 * j + 1]);
      d2p[j] = pack2h(s2.f[2 * j], s2.f[2 * j + 1]);
    }
    FragH pb;
#pragma unroll
    for (int j = 0; j < 4; ++j) {
      unsigned int o1 = (unsigned int)__shfl_xor((int)d1p[j], 16, 32);
      unsigned int o2 = (unsigned int)__shfl_xor((int)d2p[j], 16, 32);
      pb.u[j]     = g ? o2     : d1p[j];
      pb.u[4 + j] = g ? d2p[j] : o1;
    }
    // out^T += V^T(16e x 32k) x P^T(32k x 16q)
#pragma unroll
    for (int t = 0; t < 4; ++t)
      acc[t].v = wmma_f16(frag_a_h(Vl, 32, t * 16, 0), pb.v, acc[t].v);
    cur ^= 1;
  }

  float lsum = lpart + __shfl_xor(lpart, 16, 32);
  float inv = 1.f / lsum;
  _Float16* orow = attn + ((size_t)(b * NPp + qbase + nl)) * (Hh * Ee) + h * 64 + 8 * g;
#pragma unroll
  for (int t = 0; t < 4; ++t) {
    Pack8 op;
#pragma unroll
    for (int r = 0; r < 8; ++r) op.h[r] = (_Float16)(acc[t].f[r] * inv);
    *(f4*)(orow + t * 16) = op.q;
  }
}

// ---------------------------------------------------------------------------
// 7) output projection: A = woT [64][512], B = attn (N=rows) -> D[m=o][n=row]
__global__ __launch_bounds__(128) void k_oproj(
    const _Float16* __restrict__ attn, const _Float16* __restrict__ woT,
    const float* __restrict__ bo, float* __restrict__ out)
{
  int wave = blockIdx.x * 4 + (threadIdx.x >> 5);   // 1024 waves
  int rowbase = wave * 16;                          // over 16384 rows
  int l = lane_id(); int g = l >> 4; int nl = l & 15;
  AccF c[4];
#pragma unroll
  for (int t = 0; t < 4; ++t)
#pragma unroll
    for (int r = 0; r < 8; ++r) c[t].f[r] = 0.f;
  for (int rb = 0; rb < 16; ++rb) {
    v16h bf = frag_b_h(attn, Hh * Ee, rowbase, rb * 32);
#pragma unroll
    for (int t = 0; t < 4; ++t)
      c[t].v = wmma_f16(frag_a_h(woT, Hh * Ee, t * 16, rb * 32), bf, c[t].v);
  }
  float* orow = out + (size_t)(rowbase + nl) * 64 + 8 * g;
#pragma unroll
  for (int t = 0; t < 4; ++t) {
    const float* bp = bo + t * 16 + 8 * g;
#pragma unroll
    for (int r = 0; r < 8; ++r) c[t].f[r] += bp[r];
    *(f4*)(orow + t * 16)     = c[t].q[0];
    *(f4*)(orow + t * 16 + 4) = c[t].q[1];
  }
}

// ---------------------------------------------------------------------------
extern "C" void kernel_launch(void* const* d_in, const int* in_sizes, int n_in,
                              void* d_out, int out_size, void* d_ws, size_t ws_size,
                              hipStream_t stream) {
  const float* vox   = (const float*)d_in[0];
  const float* desc  = (const float*)d_in[1];
  const float* convw = (const float*)d_in[2];
  const float* convb = (const float*)d_in[3];
  const float* lng   = (const float*)d_in[4];
  const float* lnb   = (const float*)d_in[5];
  const float* pose  = (const float*)d_in[6];
  const float* wq    = (const float*)d_in[7];
  const float* bq    = (const float*)d_in[8];
  const float* wk    = (const float*)d_in[9];
  const float* bk    = (const float*)d_in[10];
  const float* wv    = (const float*)d_in[11];
  const float* bv    = (const float*)d_in[12];
  const float* wd    = (const float*)d_in[13];
  const float* bd    = (const float*)d_in[14];
  const float* alpha = (const float*)d_in[15];
  const float* wo    = (const float*)d_in[16];
  const float* bo    = (const float*)d_in[17];
  float* out = (float*)d_out;

  size_t off = 0;
  auto alloc = [&](size_t bytes) -> void* {
    void* p = (char*)d_ws + off;
    off += (bytes + 255) & ~(size_t)255;
    return p;
  };
  _Float16* wqT   = (_Float16*)alloc((size_t)512 * 64 * 2);
  _Float16* wkT   = (_Float16*)alloc((size_t)512 * 64 * 2);
  _Float16* wvT   = (_Float16*)alloc((size_t)512 * 64 * 2);
  _Float16* wdT   = (_Float16*)alloc((size_t)64 * 256 * 2);
  _Float16* woT   = (_Float16*)alloc((size_t)64 * 512 * 2);
  _Float16* vemb  = (_Float16*)alloc((size_t)Bb * NPp * Ee * 2);
  _Float16* des   = (_Float16*)alloc((size_t)Bb * SDd * Ee * 2);
  _Float16* Qh    = (_Float16*)alloc((size_t)Bb * Hh * NPp * Ee * 2);
  float*    q2    = (float*)   alloc((size_t)Bb * Hh * NPp * 4);
  _Float16* Kh    = (_Float16*)alloc((size_t)Bb * Hh * SDd * Ee * 2);
  float*    k2    = (float*)   alloc((size_t)Bb * Hh * SDd * 4);
  _Float16* VTh   = (_Float16*)alloc((size_t)Bb * Hh * Ee * SDd * 2);
  _Float16* attnb = (_Float16*)alloc((size_t)Bb * NPp * Hh * Ee * 2);
  (void)ws_size; (void)in_sizes; (void)n_in; (void)out_size;

  k_transpose<<<128, 256, 0, stream>>>(wq, wqT, 64, 512);
  k_transpose<<<128, 256, 0, stream>>>(wk, wkT, 64, 512);
  k_transpose<<<128, 256, 0, stream>>>(wv, wvT, 64, 512);
  k_transpose<<<64,  256, 0, stream>>>(wd, wdT, 256, 64);
  k_transpose<<<128, 256, 0, stream>>>(wo, woT, 512, 64);
  k_patch_embed<<<Bb * NPp, 64, 0, stream>>>(vox, convw, convb, lng, lnb, pose, vemb);
  k_desemb<<<64, 128, 0, stream>>>(desc, wdT, bd, des);
  k_qproj<<<2048, 128, 0, stream>>>(vemb, wqT, bq, Qh, q2);
  k_kvproj<<<512, 128, 0, stream>>>(des, wkT, bk, wvT, bv, Kh, k2, VTh);
  k_attn<<<2048, 128, 0, stream>>>(Qh, q2, Kh, k2, VTh, alpha, attnb);
  k_oproj<<<256, 128, 0, stream>>>(attnb, woT, bo, out);
}
